// MAE_46359876993329
// MI455X (gfx1250) — matrix operand order repaired and checked by reading, not compiled, not run on hardware
//
#include <hip/hip_runtime.h>
#include <hip/hip_bf16.h>
#include <math.h>

// ---------------------------------------------------------------- constants
#define Bn    4096
#define Nn    28
#define NDn   3
#define NMn   14
#define EDn   512
#define DDn   256
#define Hn    8
#define DHn   64
#define EMLPn 2048
#define DMLPn 1024
#define MENC  (Bn * NMn)        // 57344
#define MDEC  (Bn * Nn)         // 114688

typedef __attribute__((ext_vector_type(16))) __bf16 v16bf;
typedef __attribute__((ext_vector_type(8)))  float  v8f;

// ---------------------------------------------------------------- GEMM (WMMA)
// C[M,N] = act( A[M,K] @ W[K,N] + bias (+ Res) )
// block = 256 threads = 8 waves; block tile 128(M) x 64(N); K-step 64.
// waves: 4(M) x 2(N); each wave computes a 32x32 register tile =
// 4 x v_wmma_f32_16x16x32_bf16 accumulators; 8 WMMAs per barrier pair.
// EPI: 0 = bias only, 1 = bias + residual, 2 = bias + tanh-GELU
template <int EPI>
__global__ __launch_bounds__(256) void gemm_bf16_wmma(
    const float* __restrict__ A, const float* __restrict__ W,
    const float* __restrict__ bias, const float* __restrict__ Res,
    float* __restrict__ Out, int M, int N, int K)
{
    __shared__ __bf16 As[128][64];   // [m][k]   16 KB
    __shared__ __bf16 Bs[64][64];    // [n][k]    8 KB (transposed stage)

    const int m0   = blockIdx.x * 128;
    const int n0   = blockIdx.y * 64;
    const int t    = threadIdx.x;
    const int lane = t & 31;
    const int wave = t >> 5;
    const int wm   = wave >> 1;      // 0..3  -> 32-row subtile
    const int wn   = wave & 1;       // 0..1  -> 32-col subtile

    // staging assignments
    const int aq = t & 15;           // A: k-quad (k = aq*4)
    const int ar = t >> 4;           // A: row base (0..15), +16 per iter
    const int bn = t & 63;           // B: column n
    const int bk = t >> 6;           // B: k-group (16 k's each)

    v8f acc[2][2] = {};

    for (int k0 = 0; k0 < K; k0 += 64) {
        // ---- stage A tile 128x64: float4 loads -> packed bf16 b64 stores
#pragma unroll
        for (int rr = 0; rr < 8; ++rr) {
            const int row = ar + rr * 16;
            const float4 f = *(const float4*)&A[(size_t)(m0 + row) * K + (k0 + aq * 4)];
            union { __bf16 h[4]; uint2 v; } pk;
            pk.h[0] = (__bf16)f.x; pk.h[1] = (__bf16)f.y;
            pk.h[2] = (__bf16)f.z; pk.h[3] = (__bf16)f.w;
            *(uint2*)&As[row][aq * 4] = pk.v;
        }
        // ---- stage B tile 64x64 transposed: coalesced rows of W -> Bs[n][k]
        {
            union { __bf16 h[16]; uint4 v[2]; } pb;
#pragma unroll
            for (int j = 0; j < 16; ++j)
                pb.h[j] = (__bf16)W[(size_t)(k0 + bk * 16 + j) * N + (n0 + bn)];
            ((uint4*)&Bs[bn][bk * 16])[0] = pb.v[0];
            ((uint4*)&Bs[bn][bk * 16])[1] = pb.v[1];
        }
        __syncthreads();

        // ---- two k-subtiles of 32, 4 WMMAs each
#pragma unroll
        for (int kk = 0; kk < 64; kk += 32) {
            v16bf af[2], bfv[2];
#pragma unroll
            for (int mi = 0; mi < 2; ++mi) {
                const __bf16* ap =
                    &As[wm * 32 + mi * 16 + (lane & 15)][kk + ((lane >> 4) << 4)];
#pragma unroll
                for (int j = 0; j < 16; ++j) af[mi][j] = ap[j];
            }
#pragma unroll
            for (int ni = 0; ni < 2; ++ni) {
                const __bf16* bp =
                    &Bs[wn * 32 + ni * 16 + (lane & 15)][kk + ((lane >> 4) << 4)];
#pragma unroll
                for (int j = 0; j < 16; ++j) bfv[ni][j] = bp[j];
            }
            acc[0][0] = __builtin_amdgcn_wmma_f32_16x16x32_bf16(
                false, af[0], false, bfv[0], (short)0, acc[0][0], false, false);
            acc[0][1] = __builtin_amdgcn_wmma_f32_16x16x32_bf16(
                false, af[0], false, bfv[1], (short)0, acc[0][1], false, false);
            acc[1][0] = __builtin_amdgcn_wmma_f32_16x16x32_bf16(
                false, af[1], false, bfv[0], (short)0, acc[1][0], false, false);
            acc[1][1] = __builtin_amdgcn_wmma_f32_16x16x32_bf16(
                false, af[1], false, bfv[1], (short)0, acc[1][1], false, false);
        }
        __syncthreads();
    }

    // ---- epilogue: C/D layout — lane l: n = l&15; vgpr r: m = r + (l>=16 ? 8 : 0)
#pragma unroll
    for (int mi = 0; mi < 2; ++mi) {
#pragma unroll
        for (int ni = 0; ni < 2; ++ni) {
            const int ng = n0 + wn * 32 + ni * 16 + (lane & 15);
            const int mb = m0 + wm * 32 + mi * 16 + ((lane >> 4) << 3);
            const float bv = bias ? bias[ng] : 0.0f;
#pragma unroll
            for (int r = 0; r < 8; ++r) {
                const int mg = mb + r;
                float v = acc[mi][ni][r] + bv;
                if (EPI == 1) v += Res[(size_t)mg * N + ng];
                if (EPI == 2) {
                    float u = v;
                    v = 0.5f * u *
                        (1.0f + tanhf(0.7978845608028654f * (u + 0.044715f * u * u * u)));
                }
                Out[(size_t)mg * N + ng] = v;
            }
        }
    }
}

// ---------------------------------------------------------------- LayerNorm
// one wave (32 lanes) per row; 8 rows per 256-thread block
__global__ __launch_bounds__(256) void layernorm_k(
    const float* __restrict__ X, float* __restrict__ Y,
    const float* __restrict__ s, const float* __restrict__ b, int Wd, int M)
{
    const int row  = blockIdx.x * 8 + (threadIdx.x >> 5);
    const int lane = threadIdx.x & 31;
    if (row >= M) return;
    const float* xr = X + (size_t)row * Wd;
    float sum = 0.0f, sq = 0.0f;
    for (int c = lane; c < Wd; c += 32) { float v = xr[c]; sum += v; sq += v * v; }
#pragma unroll
    for (int o = 16; o > 0; o >>= 1) {
        sum += __shfl_xor(sum, o, 32);
        sq  += __shfl_xor(sq,  o, 32);
    }
    const float mean = sum / (float)Wd;
    const float var  = sq / (float)Wd - mean * mean;
    const float rinv = rsqrtf(var + 1e-5f);
    float* yr = Y + (size_t)row * Wd;
    for (int c = lane; c < Wd; c += 32) yr[c] = (xr[c] - mean) * rinv * s[c] + b[c];
}

// ---------------------------------------------------------------- attention
// one wave per (batch, head); lane i handles query row i (i < n).
// qkv rows: [q(0..511) | k(512..1023) | v(1024..1535)], head h -> cols h*64..
// bias: enc uses a[b][idx[i]][idx[j]] (idx = unmasked indices), dec uses a[b][i][j]
__global__ __launch_bounds__(32) void attention_k(
    const float* __restrict__ qkv, const float* __restrict__ a,
    const int* __restrict__ idx, float* __restrict__ o, int n)
{
    const int bh = blockIdx.x;
    const int b  = bh >> 3;     // H = 8
    const int h  = bh & 7;
    __shared__ float ks[Nn][DHn];
    __shared__ float vs[Nn][DHn];
    __shared__ int   ci[Nn];
    const int lane = threadIdx.x;
    const size_t rowbase = (size_t)b * n;

    if (lane < n) ci[lane] = idx ? idx[b * n + lane] : lane;
    for (int e = lane; e < n * DHn; e += 32) {
        int j = e >> 6, d = e & 63;
        ks[j][d] = qkv[(rowbase + j) * 1536 +  512 + h * 64 + d];
        vs[j][d] = qkv[(rowbase + j) * 1536 + 1024 + h * 64 + d];
    }
    __syncthreads();

    if (lane < n) {
        const int i = lane;
        float q[DHn];
        const float* qp = &qkv[(rowbase + i) * 1536 + h * 64];
#pragma unroll
        for (int d = 0; d < DHn; ++d) q[d] = qp[d];

        const int ri = ci[i];
        const float* ar = &a[((size_t)b * Nn + ri) * Nn];
        float dots[Nn];
        float mx = -1e30f;
        for (int j = 0; j < n; ++j) {
            float dt = 0.0f;
#pragma unroll
            for (int d = 0; d < DHn; ++d) dt += q[d] * ks[j][d];
            dt = dt * 0.125f + ar[ci[j]];
            dots[j] = dt;
            mx = fmaxf(mx, dt);
        }
        float sum = 0.0f;
        for (int j = 0; j < n; ++j) { dots[j] = __expf(dots[j] - mx); sum += dots[j]; }
        const float inv = 1.0f / sum;
        float* op = &o[(rowbase + i) * 512 + h * 64];
        for (int d = 0; d < DHn; ++d) {
            float av = 0.0f;
            for (int j = 0; j < n; ++j) av += dots[j] * vs[j][d];
            op[d] = av * inv;
        }
    }
}

// ---------------------------------------------------------------- small kernels
__global__ void embed_k(const float* __restrict__ x, const int* __restrict__ uidx,
                        const float* __restrict__ We, const float* __restrict__ be,
                        const float* __restrict__ pe, float* __restrict__ out)
{
    size_t tid = (size_t)blockIdx.x * blockDim.x + threadIdx.x;   // MENC*ED
    if (tid >= (size_t)MENC * EDn) return;
    const int e = (int)(tid & (EDn - 1));
    const size_t r = tid >> 9;          // b*14 + i
    const int b = (int)(r / NMn);
    const int u = uidx[r];
    const float* xp = &x[((size_t)b * Nn + u) * NDn];
    float v = be[e] + pe[(size_t)u * EDn + e];
    v += xp[0] * We[0 * EDn + e] + xp[1] * We[1 * EDn + e] + xp[2] * We[2 * EDn + e];
    out[tid] = v;
}

__global__ void gather_masked_k(const float* __restrict__ x, const int* __restrict__ midx,
                                float* __restrict__ out_nodes, float* __restrict__ out_idx)
{
    int tid = blockIdx.x * blockDim.x + threadIdx.x;   // B*NM
    if (tid >= Bn * NMn) return;
    const int b = tid / NMn;
    const int m = midx[tid];
    out_idx[tid] = (float)m;
    const float* xp = &x[((size_t)b * Nn + m) * NDn];
    out_nodes[tid * 3 + 0] = xp[0];
    out_nodes[tid * 3 + 1] = xp[1];
    out_nodes[tid * 3 + 2] = xp[2];
}

__global__ void build_dec_k(const float* __restrict__ enc_proj, const int* __restrict__ midx,
                            const float* __restrict__ mask_token, const float* __restrict__ dpe,
                            float* __restrict__ out0, float* __restrict__ xdec)
{
    size_t tid = (size_t)blockIdx.x * blockDim.x + threadIdx.x;   // MDEC*DD
    if (tid >= (size_t)MDEC * DDn) return;
    const int c = (int)(tid & (DDn - 1));
    const size_t r = tid >> 8;          // b*28 + pos
    const int pos = (int)(r % Nn);
    const int b   = (int)(r / Nn);
    float v;
    if (pos < NMn) {
        const int m = midx[b * NMn + pos];
        v = mask_token[c] + dpe[(size_t)m * DDn + c];
    } else {
        v = enc_proj[((size_t)b * NMn + (pos - NMn)) * DDn + c];
    }
    out0[tid] = v;
    xdec[tid] = v;
}

__global__ __launch_bounds__(256) void pred_loss_k(
    const float* __restrict__ xdec, const int* __restrict__ midx,
    const float* __restrict__ x, const float* __restrict__ Wp,
    const float* __restrict__ bp, float* __restrict__ out_pred,
    float* __restrict__ acc)
{
    const int tid = blockIdx.x * 256 + threadIdx.x;   // B*NM
    float lsum = 0.0f;
    if (tid < Bn * NMn) {
        const int b = tid / NMn;
        const int m = midx[tid];
        const float* dp = &xdec[((size_t)b * Nn + m) * DDn];
        const float* xp = &x[((size_t)b * Nn + m) * NDn];
#pragma unroll
        for (int o = 0; o < 3; ++o) {
            float v = bp[o];
            for (int c = 0; c < DDn; ++c) v += dp[c] * Wp[c * 3 + o];
            out_pred[tid * 3 + o] = v;
            const float d = v - xp[o];
            lsum += d * d;
        }
    }
    __shared__ float red[256];
    red[threadIdx.x] = lsum;
    __syncthreads();
    for (int s = 128; s > 0; s >>= 1) {
        if (threadIdx.x < s) red[threadIdx.x] += red[threadIdx.x + s];
        __syncthreads();
    }
    if (threadIdx.x == 0) atomicAdd(acc, red[0]);
}

__global__ void zero_acc_k(float* acc) { acc[0] = 0.0f; }
__global__ void finalize_loss_k(const float* acc, float* out_loss) {
    out_loss[0] = acc[0] / (float)(Bn * NMn * NDn);
}

// ---------------------------------------------------------------- launch
static inline void launch_gemm(int epi, const float* A, const float* W,
                               const float* bias, const float* Res, float* Out,
                               int M, int N, int K, hipStream_t s)
{
    dim3 g(M / 128, N / 64);
    if (epi == 0)      gemm_bf16_wmma<0><<<g, 256, 0, s>>>(A, W, bias, Res, Out, M, N, K);
    else if (epi == 1) gemm_bf16_wmma<1><<<g, 256, 0, s>>>(A, W, bias, Res, Out, M, N, K);
    else               gemm_bf16_wmma<2><<<g, 256, 0, s>>>(A, W, bias, Res, Out, M, N, K);
}

extern "C" void kernel_launch(void* const* d_in, const int* in_sizes, int n_in,
                              void* d_out, int out_size, void* d_ws, size_t ws_size,
                              hipStream_t stream)
{
    (void)in_sizes; (void)n_in; (void)out_size; (void)ws_size;
    const float* x        = (const float*)d_in[0];
    const float* a        = (const float*)d_in[1];
    const int*   midx     = (const int*)d_in[2];
    const int*   uidx     = (const int*)d_in[3];
    const float* W_emb    = (const float*)d_in[4];
    const float* b_emb    = (const float*)d_in[5];
    const float* pos_emb  = (const float*)d_in[6];
    const float* e_ln1_s  = (const float*)d_in[7];
    const float* e_ln1_b  = (const float*)d_in[8];
    const float* e_Wqkv   = (const float*)d_in[9];
    const float* e_Wo     = (const float*)d_in[10];
    const float* e_bo     = (const float*)d_in[11];
    const float* e_ln2_s  = (const float*)d_in[12];
    const float* e_ln2_b  = (const float*)d_in[13];
    const float* e_W1     = (const float*)d_in[14];
    const float* e_b1     = (const float*)d_in[15];
    const float* e_W2     = (const float*)d_in[16];
    const float* e_b2     = (const float*)d_in[17];
    const float* e2d_W    = (const float*)d_in[18];
    const float* e2d_b    = (const float*)d_in[19];
    const float* mask_tok = (const float*)d_in[20];
    const float* dpe      = (const float*)d_in[21];
    const float* d_ln1_s  = (const float*)d_in[22];
    const float* d_ln1_b  = (const float*)d_in[23];
    const float* d_Wqkv   = (const float*)d_in[24];
    const float* d_Wo     = (const float*)d_in[25];
    const float* d_bo     = (const float*)d_in[26];
    const float* d_ln2_s  = (const float*)d_in[27];
    const float* d_ln2_b  = (const float*)d_in[28];
    const float* d_W1     = (const float*)d_in[29];
    const float* d_b1     = (const float*)d_in[30];
    const float* d_W2     = (const float*)d_in[31];
    const float* d_b2     = (const float*)d_in[32];
    const float* pred_W   = (const float*)d_in[33];
    const float* pred_b   = (const float*)d_in[34];

    float* out = (float*)d_out;
    float* ws  = (float*)d_ws;

    // workspace layout (floats)
    float* x_enc = ws;                                        // MENC*ED  = 29360128
    float* tmp1  = x_enc + (size_t)MENC * EDn;                // MDEC*512 = 58720256
    float* tmp2  = tmp1 + (size_t)MDEC * 512;                 // MDEC*1536= 176160768
    float* x_dec = tmp2 + (size_t)MDEC * 1536;                // MDEC*DD  = 29360128
    float* acc   = x_dec + (size_t)MDEC * DDn;                // 1

    // output layout (floats, reference tuple order)
    float* out_dec  = out;                                    // B*28*256
    float* out_node = out_dec  + (size_t)MDEC * DDn;          // B*14*3
    float* out_idx  = out_node + (size_t)Bn * NMn * NDn;      // B*14
    float* out_pred = out_idx  + (size_t)Bn * NMn;            // B*14*3
    float* out_loss = out_pred + (size_t)Bn * NMn * NDn;      // 1

    zero_acc_k<<<1, 1, 0, stream>>>(acc);

    // tokens = x@W_emb + b_emb + pos_emb, gathered at unmasked positions
    {
        size_t n = (size_t)MENC * EDn;
        embed_k<<<(unsigned)((n + 255) / 256), 256, 0, stream>>>(x, uidx, W_emb, b_emb, pos_emb, x_enc);
    }
    gather_masked_k<<<(Bn * NMn + 255) / 256, 256, 0, stream>>>(x, midx, out_node, out_idx);

    // ---------------- encoder: 2 layers, M = 57344, width 512
    for (int i = 0; i < 2; ++i) {
        const float* Wqkv = e_Wqkv + (size_t)i * EDn * 1536;
        const float* Wo   = e_Wo   + (size_t)i * 512 * EDn;
        const float* W1   = e_W1   + (size_t)i * EDn * EMLPn;
        const float* W2   = e_W2   + (size_t)i * EMLPn * EDn;

        layernorm_k<<<MENC / 8, 256, 0, stream>>>(x_enc, tmp1, e_ln1_s + i * EDn, e_ln1_b + i * EDn, EDn, MENC);
        launch_gemm(0, tmp1, Wqkv, nullptr, nullptr, tmp2, MENC, 1536, EDn, stream);
        attention_k<<<Bn * Hn, 32, 0, stream>>>(tmp2, a, uidx, tmp1, NMn);
        launch_gemm(1, tmp1, Wo, e_bo + i * EDn, x_enc, x_enc, MENC, EDn, 512, stream);
        layernorm_k<<<MENC / 8, 256, 0, stream>>>(x_enc, tmp1, e_ln2_s + i * EDn, e_ln2_b + i * EDn, EDn, MENC);
        launch_gemm(2, tmp1, W1, e_b1 + i * EMLPn, nullptr, tmp2, MENC, EMLPn, EDn, stream);
        launch_gemm(1, tmp2, W2, e_b2 + i * EDn, x_enc, x_enc, MENC, EDn, EMLPn, stream);
    }

    // enc -> dec projection (into tmp1), then assemble decoder tokens
    launch_gemm(0, x_enc, e2d_W, e2d_b, nullptr, tmp1, MENC, DDn, EDn, stream);
    {
        size_t n = (size_t)MDEC * DDn;
        build_dec_k<<<(unsigned)((n + 255) / 256), 256, 0, stream>>>(tmp1, midx, mask_tok, dpe, out_dec, x_dec);
    }

    // ---------------- decoder: 1 layer, M = 114688, width 256
    {
        layernorm_k<<<MDEC / 8, 256, 0, stream>>>(x_dec, tmp1, d_ln1_s, d_ln1_b, DDn, MDEC);
        launch_gemm(0, tmp1, d_Wqkv, nullptr, nullptr, tmp2, MDEC, 1536, DDn, stream);
        attention_k<<<Bn * Hn, 32, 0, stream>>>(tmp2, a, nullptr, tmp1, Nn);
        launch_gemm(1, tmp1, d_Wo, d_bo, x_dec, x_dec, MDEC, DDn, 512, stream);
        layernorm_k<<<MDEC / 8, 256, 0, stream>>>(x_dec, tmp1, d_ln2_s, d_ln2_b, DDn, MDEC);
        launch_gemm(2, tmp1, d_W1, d_b1, nullptr, tmp2, MDEC, DMLPn, DDn, stream);
        launch_gemm(1, tmp2, d_W2, d_b2, x_dec, x_dec, MDEC, DDn, DMLPn, stream);
    }

    // prediction head + loss
    pred_loss_k<<<(Bn * NMn + 255) / 256, 256, 0, stream>>>(x_dec, midx, x, pred_W, pred_b, out_pred, acc);
    finalize_loss_k<<<1, 1, 0, stream>>>(acc, out_loss);
}
